// MultiHeadLatentAttention_37349035606740
// MI455X (gfx1250) — matrix operand order repaired and checked by reading, not compiled, not run on hardware
//
#include <hip/hip_runtime.h>

// Problem constants (fixed by the reference)
#define Bv  2
#define Sv  2048
#define Dv  2048
#define Hv  16
#define DKv 128
#define Lv  512

typedef __attribute__((ext_vector_type(16))) __bf16        v16bf;
typedef __attribute__((ext_vector_type(8)))  float         v8f;
typedef __attribute__((ext_vector_type(4)))  unsigned int  u32x4;
typedef __attribute__((ext_vector_type(2)))  unsigned int  u32x2;
typedef __attribute__((ext_vector_type(4)))  float         f32x4;

union FragBF {            // 32 bytes: one 16-bit WMMA operand (8 VGPRs)
    v16bf v;
    u32x4 u[2];
};

__device__ __forceinline__ unsigned short f32_bf16(float f) {
    unsigned int u = __float_as_uint(f);
    u += 0x7FFFu + ((u >> 16) & 1u);          // round-to-nearest-even
    return (unsigned short)(u >> 16);
}

// ---------------------------------------------------------------------------
// fp32 -> bf16 bulk conversion (vectorized, 8B stores)
// ---------------------------------------------------------------------------
__global__ __launch_bounds__(256) void f32_to_bf16_kernel(
    const float* __restrict__ in, unsigned short* __restrict__ out, int n4)
{
    int i = blockIdx.x * 256 + threadIdx.x;
    if (i >= n4) return;
    f32x4 f = *(const f32x4*)(in + (size_t)i * 4);
    u32x2 o;
    o.x = (unsigned int)f32_bf16(f.x) | ((unsigned int)f32_bf16(f.y) << 16);
    o.y = (unsigned int)f32_bf16(f.z) | ((unsigned int)f32_bf16(f.w) << 16);
    *(u32x2*)(out + (size_t)i * 4) = o;
}

// ---------------------------------------------------------------------------
// Generic batched bf16 GEMM:  C[g] = A[g] @ B[g] + bias[g]
//   A: [M,K] row-major bf16, B: [K,N] row-major bf16, bias: [N] fp32
//   Tile: BM=128 BN=128 BK=32, 256 threads (8 waves).
//   Waves in a 4x2 grid; each wave owns a 32x64 C block = 2x4 WMMA tiles
//   (8 v_wmma per k-step against 6 fragment loads).
//   B tile staged N-major in LDS via register 2x2 pair-transpose so every
//   B fragment is two contiguous ds_load_b128 per the ISA operand layout.
//   Requires M%128==0, N%128==0, K%32==0 (true for every call we make).
// ---------------------------------------------------------------------------
template <bool OUT_BF16>
__global__ __launch_bounds__(256) void gemm_bf16(
    const unsigned short* __restrict__ A,
    const unsigned short* __restrict__ Bm,
    const float*          __restrict__ bias,
    void*                 __restrict__ Cout,
    int M, int N, int K,
    long long sA, long long sB, long long sBias, long long sC)
{
    __shared__ __align__(16) unsigned short As[128 * 32];   // [m][k]
    __shared__ __align__(16) unsigned short Bs[128 * 32];   // [n][k] (transposed)

    const int t    = threadIdx.x;
    const int lane = t & 31;
    const int w    = t >> 5;
    const int lrow = lane & 15;
    const int lhi  = lane >> 4;
    const int bz   = blockIdx.z;

    A    += (size_t)bz * sA;
    Bm   += (size_t)bz * sB;
    bias += (size_t)bz * sBias;

    const int n0 = blockIdx.x * 128;
    const int m0 = blockIdx.y * 128;

    const int m_wave = (w >> 1) * 32;      // 4 wave-rows of 32
    const int n_wave = (w & 1) * 64;       // 2 wave-cols of 64

    v8f acc[2][4] = {};

    const int KT = K >> 5;
    for (int kt = 0; kt < KT; ++kt) {
        // --- stage A tile (row-major copy, 16B chunks) ---
#pragma unroll
        for (int i = 0; i < 2; ++i) {
            int c   = i * 256 + t;            // 512 chunks of 8 elems
            int row = c >> 2;
            int off = (c & 3) * 8;
            *(u32x4*)&As[row * 32 + off] =
                *(const u32x4*)&A[(size_t)(m0 + row) * K + kt * 32 + off];
        }
        // --- stage B tile transposed via 2(k)x4(n) register pair-transpose ---
#pragma unroll
        for (int i = 0; i < 2; ++i) {
            int sb = i * 256 + t;             // 512 sub-blocks
            int k  = (sb >> 5) * 2;
            int n  = (sb & 31) * 4;
            u32x2 d0 = *(const u32x2*)&Bm[(size_t)(kt * 32 + k + 0) * N + n0 + n];
            u32x2 d1 = *(const u32x2*)&Bm[(size_t)(kt * 32 + k + 1) * N + n0 + n];
            *(unsigned int*)&Bs[(n + 0) * 32 + k] = (d0.x & 0xFFFFu)     | (d1.x << 16);
            *(unsigned int*)&Bs[(n + 1) * 32 + k] = (d0.x >> 16)         | (d1.x & 0xFFFF0000u);
            *(unsigned int*)&Bs[(n + 2) * 32 + k] = (d0.y & 0xFFFFu)     | (d1.y << 16);
            *(unsigned int*)&Bs[(n + 3) * 32 + k] = (d0.y >> 16)         | (d1.y & 0xFFFF0000u);
        }
        __syncthreads();

        // A fragments: lane m=lrow, halves = K {koff..koff+7, koff+16..koff+23}
        FragBF a[2];
#pragma unroll
        for (int tm = 0; tm < 2; ++tm) {
            int arow = m_wave + tm * 16 + lrow;
            int koff = lhi * 8;
            a[tm].u[0] = *(const u32x4*)&As[arow * 32 + koff];
            a[tm].u[1] = *(const u32x4*)&As[arow * 32 + koff + 16];
        }
#pragma unroll
        for (int tn = 0; tn < 4; ++tn) {
            FragBF bf;
            int bcol = n_wave + tn * 16 + lrow;
            int bk   = lhi * 8;               // halves = K {bk..bk+7, bk+16..bk+23}
            bf.u[0] = *(const u32x4*)&Bs[bcol * 32 + bk];
            bf.u[1] = *(const u32x4*)&Bs[bcol * 32 + bk + 16];
#pragma unroll
            for (int tm = 0; tm < 2; ++tm) {
                acc[tm][tn] = __builtin_amdgcn_wmma_f32_16x16x32_bf16(
                    false, a[tm].v, false, bf.v, (short)0, acc[tm][tn], false, false);
            }
        }
        __syncthreads();
    }

    // Epilogue: C layout -> row = r + 8*lhi, col = lrow (per 16-col tile)
#pragma unroll
    for (int tn = 0; tn < 4; ++tn) {
        const int   col  = n0 + n_wave + tn * 16 + lrow;
        const float bval = bias[col];
#pragma unroll
        for (int tm = 0; tm < 2; ++tm) {
#pragma unroll
            for (int r = 0; r < 8; ++r) {
                int row = m0 + m_wave + tm * 16 + r + 8 * lhi;
                float v = acc[tm][tn][r] + bval;
                size_t idx = (size_t)bz * sC + (size_t)row * N + col;
                if (OUT_BF16) ((unsigned short*)Cout)[idx] = f32_bf16(v);
                else          ((float*)Cout)[idx] = v;
            }
        }
    }
}

// ---------------------------------------------------------------------------
// Fused flash attention (one pass, scores never hit HBM).
//   Q:    [B,S,D] bf16 (heads packed, head h at column h*DK)
//   K,V:  [H,B,S,DK] bf16
//   Out:  [B,S,D] bf16
// Block = 256 threads (8 waves) handles 128 query rows of one (b,h), all
// waves sharing each 32-key K/V LDS tile. K is staged row-major (QK^T
// B-fragments are contiguous b128s), V is staged transposed with a 2x4
// register pair-transpose (PV B-fragments contiguous). Online softmax runs
// in the WMMA C-layout with 16-lane shfl_xor row reductions.
// ---------------------------------------------------------------------------
__global__ __launch_bounds__(256) void mla_attn(
    const unsigned short* __restrict__ Q,
    const unsigned short* __restrict__ Km,
    const unsigned short* __restrict__ Vm,
    unsigned short*       __restrict__ O)
{
    __shared__ __align__(16) unsigned short Ks[32 * 128];   // [key][dk]
    __shared__ __align__(16) unsigned short Vs[128 * 32];   // [dk][key]
    __shared__ __align__(16) unsigned short Ps[8][16 * 32]; // per-wave P tile

    const int t    = threadIdx.x;
    const int lane = t & 31;
    const int w    = t >> 5;
    const int lrow = lane & 15;
    const int lhi  = lane >> 4;
    const int h    = blockIdx.y;
    const int b    = blockIdx.z;
    const int q0   = blockIdx.x * 128 + w * 16;

    // Preload the wave's 16x128 Q block as 4 A-fragments (dk chunks of 32)
    FragBF qf[4];
#pragma unroll
    for (int c = 0; c < 4; ++c) {
        size_t base = ((size_t)(b * Sv + q0 + lrow)) * Dv + h * DKv + c * 32 + lhi * 8;
        qf[c].u[0] = *(const u32x4*)&Q[base];
        qf[c].u[1] = *(const u32x4*)&Q[base + 16];
    }

    v8f   o[8] = {};
    float mrow[8], lsum[8];
#pragma unroll
    for (int r = 0; r < 8; ++r) { mrow[r] = -3.0e38f; lsum[r] = 0.0f; }

    const unsigned short* KH = Km + ((size_t)(h * Bv + b)) * Sv * DKv;
    const unsigned short* VH = Vm + ((size_t)(h * Bv + b)) * Sv * DKv;
    const float inv_scale = 0.088388347648318447f;   // 1/sqrt(128)

    for (int kb = 0; kb < Sv / 32; ++kb) {
        const unsigned short* Kt = KH + (size_t)kb * 32 * DKv;
        const unsigned short* Vt = VH + (size_t)kb * 32 * DKv;
        // K tile: straight 16B copies (layout identical)
#pragma unroll
        for (int i = 0; i < 2; ++i) {
            int c = i * 256 + t;                       // 512 chunks of 8 elems
            *(u32x4*)&Ks[c * 8] = *(const u32x4*)&Kt[(size_t)c * 8];
        }
        // V tile transposed: Vs[dk][key], 2(key)x4(dk) pair-transpose
#pragma unroll
        for (int i = 0; i < 2; ++i) {
            int sb  = i * 256 + t;                     // 512 sub-blocks
            int key = (sb >> 5) * 2;
            int dk  = (sb & 31) * 4;
            u32x2 d0 = *(const u32x2*)&Vt[(size_t)(key + 0) * DKv + dk];
            u32x2 d1 = *(const u32x2*)&Vt[(size_t)(key + 1) * DKv + dk];
            *(unsigned int*)&Vs[(dk + 0) * 32 + key] = (d0.x & 0xFFFFu) | (d1.x << 16);
            *(unsigned int*)&Vs[(dk + 1) * 32 + key] = (d0.x >> 16)     | (d1.x & 0xFFFF0000u);
            *(unsigned int*)&Vs[(dk + 2) * 32 + key] = (d0.y & 0xFFFFu) | (d1.y << 16);
            *(unsigned int*)&Vs[(dk + 3) * 32 + key] = (d0.y >> 16)     | (d1.y & 0xFFFF0000u);
        }
        __syncthreads();

        // --- scores S = Q @ K^T : 16 queries x 32 keys, contraction dk=128 ---
        v8f s0 = {}, s1 = {};
#pragma unroll
        for (int c = 0; c < 4; ++c) {
            FragBF b0, b1;
            int koff = c * 32 + lhi * 8;
            b0.u[0] = *(const u32x4*)&Ks[lrow * DKv + koff];
            b0.u[1] = *(const u32x4*)&Ks[lrow * DKv + koff + 16];
            b1.u[0] = *(const u32x4*)&Ks[(16 + lrow) * DKv + koff];
            b1.u[1] = *(const u32x4*)&Ks[(16 + lrow) * DKv + koff + 16];
            s0 = __builtin_amdgcn_wmma_f32_16x16x32_bf16(
                false, qf[c].v, false, b0.v, (short)0, s0, false, false);
            s1 = __builtin_amdgcn_wmma_f32_16x16x32_bf16(
                false, qf[c].v, false, b1.v, (short)0, s1, false, false);
        }

        // --- online softmax per row (rows live per C-layout VGPR slot) ---
#pragma unroll
        for (int r = 0; r < 8; ++r) {
            float a0 = s0[r] * inv_scale;
            float a1 = s1[r] * inv_scale;
            float mx = fmaxf(a0, a1);
#pragma unroll
            for (int m = 1; m < 16; m <<= 1) mx = fmaxf(mx, __shfl_xor(mx, m, 32));
            float mnew  = fmaxf(mrow[r], mx);
            float alpha = __expf(mrow[r] - mnew);
            float p0    = __expf(a0 - mnew);
            float p1    = __expf(a1 - mnew);
            float rs    = p0 + p1;
#pragma unroll
            for (int m = 1; m < 16; m <<= 1) rs += __shfl_xor(rs, m, 32);
            lsum[r] = lsum[r] * alpha + rs;
            mrow[r] = mnew;
#pragma unroll
            for (int tt = 0; tt < 8; ++tt) o[tt][r] *= alpha;
            int prow = r + 8 * lhi;
            Ps[w][prow * 32 + lrow]      = f32_bf16(p0);
            Ps[w][prow * 32 + 16 + lrow] = f32_bf16(p1);
        }

        // --- O += P @ V (wave-local P round-trip: C-layout -> A-layout) ---
        FragBF ap;
        ap.u[0] = *(const u32x4*)&Ps[w][lrow * 32 + lhi * 8];
        ap.u[1] = *(const u32x4*)&Ps[w][lrow * 32 + lhi * 8 + 16];
#pragma unroll
        for (int tt = 0; tt < 8; ++tt) {
            FragBF bv;
            int dk = tt * 16 + lrow;
            bv.u[0] = *(const u32x4*)&Vs[dk * 32 + lhi * 8];
            bv.u[1] = *(const u32x4*)&Vs[dk * 32 + lhi * 8 + 16];
            o[tt] = __builtin_amdgcn_wmma_f32_16x16x32_bf16(
                false, ap.v, false, bv.v, (short)0, o[tt], false, false);
        }
        __syncthreads();
    }

    // Finalize: divide by running sum, emit bf16 [B,S,D]
#pragma unroll
    for (int r = 0; r < 8; ++r) {
        const float rinv = 1.0f / lsum[r];
        const int   row  = q0 + r + 8 * lhi;
#pragma unroll
        for (int tt = 0; tt < 8; ++tt) {
            float v = o[tt][r] * rinv;
            O[((size_t)(b * Sv) + row) * Dv + h * DKv + tt * 16 + lrow] = f32_bf16(v);
        }
    }
}

// ---------------------------------------------------------------------------
extern "C" void kernel_launch(void* const* d_in, const int* in_sizes, int n_in,
                              void* d_out, int out_size, void* d_ws, size_t ws_size,
                              hipStream_t stream)
{
    (void)in_sizes; (void)n_in; (void)out_size; (void)ws_size;
    const float* query = (const float*)d_in[0];
    const float* key   = (const float*)d_in[1];
    // d_in[2] = value : unused by the reference (MLA reconstructs V from latent)
    const float* Wq  = (const float*)d_in[3];
    const float* bq  = (const float*)d_in[4];
    const float* Wc  = (const float*)d_in[5];
    const float* bc  = (const float*)d_in[6];
    const float* WkA = (const float*)d_in[7];
    const float* bkA = (const float*)d_in[8];
    const float* WkB = (const float*)d_in[9];
    const float* bkB = (const float*)d_in[10];
    const float* WvA = (const float*)d_in[11];
    const float* bvA = (const float*)d_in[12];
    const float* WvB = (const float*)d_in[13];
    const float* bvB = (const float*)d_in[14];
    const float* Wo  = (const float*)d_in[15];
    const float* bo  = (const float*)d_in[16];

    const size_t MS = (size_t)Bv * Sv;                 // 4096 rows everywhere

    char*  ws  = (char*)d_ws;
    size_t off = 0;
    auto alloc = [&](size_t elems) -> unsigned short* {
        unsigned short* p = (unsigned short*)(ws + off);
        off += (elems * sizeof(unsigned short) + 255) & ~(size_t)255;
        return p;
    };
    unsigned short* qb   = alloc(MS * Dv);             // bf16 query  (reused as attn-out)
    unsigned short* kb   = alloc(MS * Dv);             // bf16 key
    unsigned short* Wqb  = alloc((size_t)Dv * Dv);
    unsigned short* Wcb  = alloc((size_t)Dv * Lv);
    unsigned short* WkAb = alloc((size_t)Hv * Lv * DKv);
    unsigned short* WkBb = alloc((size_t)Hv * DKv * DKv);
    unsigned short* WvAb = alloc((size_t)Hv * Lv * DKv);
    unsigned short* WvBb = alloc((size_t)Hv * DKv * DKv);
    unsigned short* Wob  = alloc((size_t)Dv * Dv);
    unsigned short* Qp   = alloc(MS * Dv);             // projected Q [B,S,H*DK]
    unsigned short* lat  = alloc(MS * Lv);             // latent [B*S, L]
    unsigned short* midb = alloc((size_t)Hv * MS * DKv); // k_mid, then reused for v_mid
    unsigned short* Kb   = alloc((size_t)Hv * MS * DKv); // [H,B,S,DK]
    unsigned short* Vb   = alloc((size_t)Hv * MS * DKv); // [H,B,S,DK]
    unsigned short* attn = qb;                         // reuse: qb dead after Q GEMM

    auto conv = [&](const float* src, unsigned short* dst, size_t n) {
        int n4 = (int)(n / 4);
        f32_to_bf16_kernel<<<dim3((n4 + 255) / 256), dim3(256), 0, stream>>>(src, dst, n4);
    };
    conv(query, qb,   MS * Dv);
    conv(key,   kb,   MS * Dv);
    conv(Wq,    Wqb,  (size_t)Dv * Dv);
    conv(Wc,    Wcb,  (size_t)Dv * Lv);
    conv(WkA,   WkAb, (size_t)Hv * Lv * DKv);
    conv(WkB,   WkBb, (size_t)Hv * DKv * DKv);
    conv(WvA,   WvAb, (size_t)Hv * Lv * DKv);
    conv(WvB,   WvBb, (size_t)Hv * DKv * DKv);
    conv(Wo,    Wob,  (size_t)Dv * Dv);

    const dim3 blk(256);
    // 1) Q = query @ Wq + bq                      [4096,2048]x[2048,2048]
    gemm_bf16<true><<<dim3(Dv / 128, MS / 128, 1), blk, 0, stream>>>(
        qb, Wqb, bq, Qp, (int)MS, Dv, Dv, 0, 0, 0, 0);
    // 2) latent = key @ Wc + bc                   [4096,2048]x[2048,512]
    gemm_bf16<true><<<dim3(Lv / 128, MS / 128, 1), blk, 0, stream>>>(
        kb, Wcb, bc, lat, (int)MS, Lv, Dv, 0, 0, 0, 0);
    // 3) k_mid[h] = latent @ WkA[h] + bkA[h]      batched over H
    gemm_bf16<true><<<dim3(DKv / 128, MS / 128, Hv), blk, 0, stream>>>(
        lat, WkAb, bkA, midb, (int)MS, DKv, Lv,
        0, (long long)Lv * DKv, DKv, (long long)MS * DKv);
    // 4) K[h] = k_mid[h] @ WkB[h] + bkB[h]
    gemm_bf16<true><<<dim3(DKv / 128, MS / 128, Hv), blk, 0, stream>>>(
        midb, WkBb, bkB, Kb, (int)MS, DKv, DKv,
        (long long)MS * DKv, (long long)DKv * DKv, DKv, (long long)MS * DKv);
    // 5) v_mid[h] = latent @ WvA[h] + bvA[h]  (reuses midb)
    gemm_bf16<true><<<dim3(DKv / 128, MS / 128, Hv), blk, 0, stream>>>(
        lat, WvAb, bvA, midb, (int)MS, DKv, Lv,
        0, (long long)Lv * DKv, DKv, (long long)MS * DKv);
    // 6) V[h] = v_mid[h] @ WvB[h] + bvB[h]
    gemm_bf16<true><<<dim3(DKv / 128, MS / 128, Hv), blk, 0, stream>>>(
        midb, WvBb, bvB, Vb, (int)MS, DKv, DKv,
        (long long)MS * DKv, (long long)DKv * DKv, DKv, (long long)MS * DKv);
    // 7) fused flash attention -> attn [B,S,D] bf16
    mla_attn<<<dim3(Sv / 128, Hv, Bv), blk, 0, stream>>>(Qp, Kb, Vb, attn);
    // 8) out = attn @ Wo + bo  (fp32 output)
    gemm_bf16<false><<<dim3(Dv / 128, MS / 128, 1), blk, 0, stream>>>(
        attn, Wob, bo, d_out, (int)MS, Dv, Dv, 0, 0, 0, 0);
}
